// CrossModalAttention_2336462209362
// MI455X (gfx1250) — matrix operand order repaired and checked by reading, not compiled, not run on hardware
//
#include <hip/hip_runtime.h>
#include <hip/hip_bf16.h>

// ---------------------------------------------------------------------------
// CrossModalAttention on MI455X (gfx1250, wave32, WMMA, async-to-LDS DMA)
//
// Math (softmax over 1 key == 1, so Q/K projections cancel):
//   per modality:  mid = kv @ Wv^T + bv
//                  X   = LN(x + mid @ Wo^T + bo)
//                  h   = gelu(X @ W1^T + b1)        (exact erf GELU)
//                  out = LN(X + h @ W2^T + b2)
// ---------------------------------------------------------------------------

typedef __bf16  v16bf __attribute__((ext_vector_type(16)));
typedef float   v8f   __attribute__((ext_vector_type(8)));

#define B_ROWS   65536
#define E_DIM    512
#define FF_DIM   1024
#define LDSTRIDE 40   // padded LDS row stride in bf16 (80 B): bank-conflict-free,
                      // keeps ds_load_b128 16-byte aligned (20 DWORDs/row)

union Frag32B { v16bf v; uint4 q[2]; };

// f32 -> bf16 weight conversion (weights are tiny; L2 resident thereafter)
__global__ void cvt_f32_bf16(const float* __restrict__ s, __bf16* __restrict__ d, int n) {
    int i = blockIdx.x * blockDim.x + threadIdx.x;
    if (i < n) d[i] = (__bf16)s[i];
}

// 64-byte cache->LDS async copy (ASYNCcnt). Both chunks share one inst_offset
// because the offset applies to BOTH the LDS and global address (ISA 10.x).
__device__ __forceinline__ void async_copy_64B(unsigned lds, unsigned long long gaddr) {
    asm volatile("global_load_async_to_lds_b128 %0, %1, off\n\t"
                 "global_load_async_to_lds_b128 %0, %1, off offset:32"
                 :: "v"(lds), "v"(gaddr) : "memory");
}
__device__ __forceinline__ void wait_async_le2() {
    asm volatile("s_wait_xcnt 0x0\n\ts_wait_asynccnt 0x2" ::: "memory");
}
__device__ __forceinline__ void wait_async_le0() {
    asm volatile("s_wait_xcnt 0x0\n\ts_wait_asynccnt 0x0" ::: "memory");
}

// ---------------------------------------------------------------------------
// Tiled GEMM:  out[r, c] = A[r, :K] @ W[c, :K]^T + bias[c]   (+ fused epilogue)
//   EPI == 0 : exact GELU, write bf16
//   EPI == 1 : + res[r,c], LayerNorm over the N=512 row, write f32
//   EPI == 2 : bias only, write bf16
// Block: 512 threads = 16 waves; tile = 64 rows x 512 cols, double-buffered LDS.
// Wave (mw,nw) in 4x4 grid: 16 rows x 128 cols = 8 WMMA accum tiles.
// ---------------------------------------------------------------------------
template<int EPI, bool AF32>
__global__ __launch_bounds__(512, 1)
void gemm_epi(const void* __restrict__ Aptr,
              const __bf16* __restrict__ W,      // [N][K] bf16
              const float*  __restrict__ bias,   // [N]
              const float*  __restrict__ res,    // [B][N] f32 (EPI==1)
              const float*  __restrict__ gam,    // [N]      (EPI==1)
              const float*  __restrict__ bet,    // [N]      (EPI==1)
              float*        __restrict__ outF,   // [B][N]   (EPI==1)
              __bf16*       __restrict__ outB,   // [B][N]   (EPI==0/2)
              int K, int N)
{
    __shared__ __bf16 sA[2][64  * LDSTRIDE];     // 10 KB (double-buffered, padded)
    __shared__ __bf16 sW[2][512 * LDSTRIDE];     // 80 KB (double-buffered, padded)
    __shared__ float  sRed[8 * 64];              //  2 KB: [0..3]=sum, [4..7]=sumsq per nw
    __shared__ float  sMean[64], sRstd[64];

    const int tid  = threadIdx.x;
    const int wave = tid >> 5;
    const int lane = tid & 31;
    const int lo   = lane & 15;          // N-col (B/C/D) or M-row (A) within tile
    const int hi   = lane >> 4;          // K-half selector
    const int mw   = wave >> 2;          // 0..3 : M sub-tile
    const int nw   = wave & 3;           // 0..3 : N sub-tile (128 cols)

    const int r0 = blockIdx.x * 64;      // global row base
    const int c0 = blockIdx.y * 512;     // global col base

    const float*  Af = (const float*)Aptr;
    const __bf16* Ab = (const __bf16*)Aptr;

    v8f acc[8];
    #pragma unroll
    for (int t = 0; t < 8; ++t)
        acc[t] = (v8f){0.f,0.f,0.f,0.f,0.f,0.f,0.f,0.f};

    // per-thread staging coordinates
    const int aRow = tid >> 3;                 // A: 64 rows, 8 threads/row
    const int aKq  = (tid & 7) * 4;            // A: 4 contiguous k per thread
    const unsigned wLds[2] = {
        (unsigned)(unsigned long long)&sW[0][tid * LDSTRIDE],
        (unsigned)(unsigned long long)&sW[1][tid * LDSTRIDE] };
    const unsigned long long wGbase =
        (unsigned long long)(const void*)(W + (size_t)(c0 + tid) * K);

    auto stageA = [&](int buf, int k0) {
        if (AF32) {
            const float4 f = *(const float4*)(Af + (size_t)(r0 + aRow) * K + k0 + aKq);
            __bf16 t4[4] = { (__bf16)f.x, (__bf16)f.y, (__bf16)f.z, (__bf16)f.w };
            *(uint2*)&sA[buf][aRow * LDSTRIDE + aKq] = *(const uint2*)t4;
        } else {
            *(uint2*)&sA[buf][aRow * LDSTRIDE + aKq] =
                *(const uint2*)(Ab + (size_t)(r0 + aRow) * K + k0 + aKq);
        }
    };

    // ---- prologue: DMA W tile 0, stage A tile 0 ----
    async_copy_64B(wLds[0], wGbase);           // 32 bf16 k-slice at k0=0
    stageA(0, 0);

    int cur = 0;
    for (int k0 = 0; k0 < K; k0 += 32) {
        const int  nxt  = cur ^ 1;
        const bool more = (k0 + 32) < K;

        // ---- prefetch next tile into the other buffer (overlaps WMMA) ----
        if (more) {
            async_copy_64B(wLds[nxt], wGbase + (size_t)(k0 + 32) * 2);
            stageA(nxt, k0 + 32);
            wait_async_le2();                  // current tile's DMA has landed
        } else {
            wait_async_le0();
        }
        __syncthreads();                       // cur tile visible to all waves

        // ---- A fragment (16x32 bf16): ISA wave32 layout ----
        // lane lo = M row; hi=0 -> K {0..7,16..23}, hi=1 -> K {8..15,24..31}
        Frag32B afr;
        const __bf16* ap = &sA[cur][(mw * 16 + lo) * LDSTRIDE];
        afr.q[0] = *(const uint4*)(ap + hi * 8);
        afr.q[1] = *(const uint4*)(ap + 16 + hi * 8);

        // ---- preload all 8 B fragments, then burst 8 independent WMMAs ----
        Frag32B bfr[8];
        #pragma unroll
        for (int tn = 0; tn < 8; ++tn) {
            // B (K x 16): lane lo = N col; hi=0 -> K {0..7,16..23}, hi=1 -> K {8..15,24..31}
            const __bf16* bp = &sW[cur][(nw * 128 + tn * 16 + lo) * LDSTRIDE + hi * 8];
            bfr[tn].q[0] = *(const uint4*)(bp);
            bfr[tn].q[1] = *(const uint4*)(bp + 16);
        }
        #pragma unroll
        for (int tn = 0; tn < 8; ++tn) {
            acc[tn] = __builtin_amdgcn_wmma_f32_16x16x32_bf16(
                false, afr.v, false, bfr[tn].v, (short)0, acc[tn], false, false);
        }
        __syncthreads();                       // all waves done reading cur
        cur = nxt;
    }

    // ------------------------- epilogues -------------------------
    if (EPI == 0 || EPI == 2) {
        #pragma unroll
        for (int tn = 0; tn < 8; ++tn) {
            const int colg = c0 + nw * 128 + tn * 16 + lo;
            const float bb = bias[colg];
            #pragma unroll
            for (int v = 0; v < 8; ++v) {
                const int rowg = r0 + mw * 16 + v + hi * 8;
                float x = acc[tn][v] + bb;
                if (EPI == 0)  // exact GELU: 0.5*x*(1+erf(x/sqrt(2)))
                    x = 0.5f * x * (1.0f + erff(x * 0.70710678118654752f));
                outB[(size_t)rowg * N + colg] = (__bf16)x;
            }
        }
    } else {
        // EPI==1: residual + LayerNorm over the full N=512 row
        float s[8], sq[8];
        #pragma unroll
        for (int v = 0; v < 8; ++v) { s[v] = 0.f; sq[v] = 0.f; }

        #pragma unroll
        for (int tn = 0; tn < 8; ++tn) {
            const int colg = c0 + nw * 128 + tn * 16 + lo;
            const float bb = bias[colg];
            #pragma unroll
            for (int v = 0; v < 8; ++v) {
                const int rowg = r0 + mw * 16 + v + hi * 8;
                float val = acc[tn][v] + bb + res[(size_t)rowg * N + colg];
                acc[tn][v] = val;
                s[v]  += val;
                sq[v] += val * val;
            }
        }
        // reduce over the 16 lanes of each half-wave (wave32, width 16)
        #pragma unroll
        for (int m = 1; m < 16; m <<= 1) {
            #pragma unroll
            for (int v = 0; v < 8; ++v) {
                s[v]  += __shfl_xor(s[v],  m, 16);
                sq[v] += __shfl_xor(sq[v], m, 16);
            }
        }
        if (lo == 0) {
            #pragma unroll
            for (int v = 0; v < 8; ++v) {
                const int rl = mw * 16 + v + hi * 8;
                sRed[nw * 64 + rl]       = s[v];
                sRed[(4 + nw) * 64 + rl] = sq[v];
            }
        }
        __syncthreads();
        if (tid < 64) {
            float su = 0.f, sk = 0.f;
            #pragma unroll
            for (int w = 0; w < 4; ++w) {
                su += sRed[w * 64 + tid];
                sk += sRed[(4 + w) * 64 + tid];
            }
            const float mean = su * (1.0f / 512.0f);
            const float var  = sk * (1.0f / 512.0f) - mean * mean;
            sMean[tid] = mean;
            sRstd[tid] = rsqrtf(var + 1e-5f);
        }
        __syncthreads();
        #pragma unroll
        for (int tn = 0; tn < 8; ++tn) {
            const int colg = c0 + nw * 128 + tn * 16 + lo;
            const float gg = gam[colg];
            const float be = bet[colg];
            #pragma unroll
            for (int v = 0; v < 8; ++v) {
                const int rl   = mw * 16 + v + hi * 8;
                const int rowg = r0 + rl;
                const float y = (acc[tn][v] - sMean[rl]) * sRstd[rl] * gg + be;
                outF[(size_t)rowg * N + colg] = y;
            }
        }
    }
}

// ---------------------------------------------------------------------------
extern "C" void kernel_launch(void* const* d_in, const int* in_sizes, int n_in,
                              void* d_out, int out_size, void* d_ws, size_t ws_size,
                              hipStream_t stream) {
    const float* text_feat = (const float*)d_in[0];
    const float* img_feat  = (const float*)d_in[1];
    // i2t: 2=Wq 3=Wk 4=Wv 5=Wo 6=bq 7=bk 8=bv 9=bo
    const float* i2t_Wv = (const float*)d_in[4];
    const float* i2t_Wo = (const float*)d_in[5];
    const float* i2t_bv = (const float*)d_in[8];
    const float* i2t_bo = (const float*)d_in[9];
    // t2i: 10..17
    const float* t2i_Wv = (const float*)d_in[12];
    const float* t2i_Wo = (const float*)d_in[13];
    const float* t2i_bv = (const float*)d_in[16];
    const float* t2i_bo = (const float*)d_in[17];
    // ffn_img: 18=W1 19=b1 20=W2 21=b2 ; ffn_text: 22..25
    const float* fi_W1 = (const float*)d_in[18];
    const float* fi_b1 = (const float*)d_in[19];
    const float* fi_W2 = (const float*)d_in[20];
    const float* fi_b2 = (const float*)d_in[21];
    const float* ft_W1 = (const float*)d_in[22];
    const float* ft_b1 = (const float*)d_in[23];
    const float* ft_W2 = (const float*)d_in[24];
    const float* ft_b2 = (const float*)d_in[25];
    // ln: 26=ln1i_g 27=ln1i_b 28=ln1t_g 29=ln1t_b 30=ln2i_g 31=ln2i_b 32=ln2t_g 33=ln2t_b
    const float* ln1i_g = (const float*)d_in[26];
    const float* ln1i_b = (const float*)d_in[27];
    const float* ln1t_g = (const float*)d_in[28];
    const float* ln1t_b = (const float*)d_in[29];
    const float* ln2i_g = (const float*)d_in[30];
    const float* ln2i_b = (const float*)d_in[31];
    const float* ln2t_g = (const float*)d_in[32];
    const float* ln2t_b = (const float*)d_in[33];

    float* out_text = (float*)d_out;                                // (B,E)
    float* out_img  = (float*)d_out + (size_t)B_ROWS * E_DIM;       // (B,E)

    // ---- workspace carve-out ----
    char* ws = (char*)d_ws;
    size_t off = 0;
    auto alloc = [&](size_t bytes) -> char* {
        size_t p = (off + 255) & ~(size_t)255;
        off = p + bytes;
        return ws + p;
    };
    const size_t EE = (size_t)E_DIM * E_DIM;
    const size_t EF = (size_t)E_DIM * FF_DIM;
    __bf16* wb_i2t_Wv = (__bf16*)alloc(EE * 2);
    __bf16* wb_i2t_Wo = (__bf16*)alloc(EE * 2);
    __bf16* wb_t2i_Wv = (__bf16*)alloc(EE * 2);
    __bf16* wb_t2i_Wo = (__bf16*)alloc(EE * 2);
    __bf16* wb_fi_W1  = (__bf16*)alloc(EF * 2);
    __bf16* wb_fi_W2  = (__bf16*)alloc(EF * 2);
    __bf16* wb_ft_W1  = (__bf16*)alloc(EF * 2);
    __bf16* wb_ft_W2  = (__bf16*)alloc(EF * 2);
    __bf16* midh = (__bf16*)alloc((size_t)B_ROWS * FF_DIM * 2);  // mid (BxE) / h (BxFF)
    float*  Xbuf = (float*) alloc((size_t)B_ROWS * E_DIM * 4);   // post-LN1 activations

    // ---- weight conversion (f32 -> bf16) ----
    auto cvt = [&](const float* s, __bf16* d, size_t n) {
        cvt_f32_bf16<<<(unsigned)((n + 255) / 256), 256, 0, stream>>>(s, d, (int)n);
    };
    cvt(i2t_Wv, wb_i2t_Wv, EE);  cvt(i2t_Wo, wb_i2t_Wo, EE);
    cvt(t2i_Wv, wb_t2i_Wv, EE);  cvt(t2i_Wo, wb_t2i_Wo, EE);
    cvt(fi_W1,  wb_fi_W1,  EF);  cvt(fi_W2,  wb_fi_W2,  EF);
    cvt(ft_W1,  wb_ft_W1,  EF);  cvt(ft_W2,  wb_ft_W2,  EF);

    const dim3 gE(B_ROWS / 64, 1);   // N = 512
    const dim3 gF(B_ROWS / 64, 2);   // N = 1024
    const dim3 blk(512);

    // ================= image path (queries text context) =================
    // mid = text @ i2t_Wv^T + bv
    gemm_epi<2, true ><<<gE, blk, 0, stream>>>(text_feat, wb_i2t_Wv, i2t_bv,
        nullptr, nullptr, nullptr, nullptr, midh, E_DIM, E_DIM);
    // X = LN1i(img + mid @ i2t_Wo^T + bo)
    gemm_epi<1, false><<<gE, blk, 0, stream>>>(midh, wb_i2t_Wo, i2t_bo,
        img_feat, ln1i_g, ln1i_b, Xbuf, nullptr, E_DIM, E_DIM);
    // h = gelu(X @ fi_W1^T + b1)
    gemm_epi<0, true ><<<gF, blk, 0, stream>>>(Xbuf, wb_fi_W1, fi_b1,
        nullptr, nullptr, nullptr, nullptr, midh, E_DIM, FF_DIM);
    // out_img = LN2i(X + h @ fi_W2^T + b2)
    gemm_epi<1, false><<<gE, blk, 0, stream>>>(midh, wb_fi_W2, fi_b2,
        Xbuf, ln2i_g, ln2i_b, out_img, nullptr, FF_DIM, E_DIM);

    // ================= text path (queries image context) =================
    gemm_epi<2, true ><<<gE, blk, 0, stream>>>(img_feat, wb_t2i_Wv, t2i_bv,
        nullptr, nullptr, nullptr, nullptr, midh, E_DIM, E_DIM);
    gemm_epi<1, false><<<gE, blk, 0, stream>>>(midh, wb_t2i_Wo, t2i_bo,
        text_feat, ln1t_g, ln1t_b, Xbuf, nullptr, E_DIM, E_DIM);
    gemm_epi<0, true ><<<gF, blk, 0, stream>>>(Xbuf, wb_ft_W1, ft_b1,
        nullptr, nullptr, nullptr, nullptr, midh, E_DIM, FF_DIM);
    gemm_epi<1, false><<<gE, blk, 0, stream>>>(midh, wb_ft_W2, ft_b2,
        Xbuf, ln2t_g, ln2t_b, out_text, nullptr, FF_DIM, E_DIM);
}
